// DecentralPlannerGATNet_75849122447969
// MI455X (gfx1250) — matrix operand order repaired and checked by reading, not compiled, not run on hardware
//
#include <hip/hip_runtime.h>

// ---------------------------------------------------------------------------
// MI455X (gfx1250) implementation of DecentralPlannerGATNet forward.
// wave32, WMMA f32_16x16x32_f16 for all dense math, f32 accumulate.
// Async global->LDS staging (ASYNCcnt) where the toolchain provides it.
// ---------------------------------------------------------------------------

typedef _Float16 half_t;
typedef __attribute__((ext_vector_type(16))) _Float16 v16h;
typedef __attribute__((ext_vector_type(8)))  _Float16 v8h;
typedef __attribute__((ext_vector_type(8)))  float    v8f;
typedef int v4i_vs __attribute__((vector_size(16)));   // matches async-LDS builtin param

#define NNODES 20000
#define NEDGES 250000
#define EAUG   (NNODES + NEDGES)   // edges + self loops

#if __has_builtin(__builtin_amdgcn_global_load_async_to_lds_b128) && \
    __has_builtin(__builtin_amdgcn_s_wait_asynccnt)
#define HAVE_ASYNC 1
#else
#define HAVE_ASYNC 0
#endif

// 16-byte global -> LDS copy; async (GLOBAL_LOAD_ASYNC_TO_LDS_B128) if available.
__device__ inline void lds_copy16(half_t* dst, const half_t* src) {
#if HAVE_ASYNC
    auto g = reinterpret_cast<__attribute__((address_space(1))) v4i_vs*>(
        reinterpret_cast<unsigned long long>(src));
    auto l = reinterpret_cast<__attribute__((address_space(3))) v4i_vs*>(
        static_cast<unsigned>(reinterpret_cast<unsigned long long>(dst)));
    __builtin_amdgcn_global_load_async_to_lds_b128(g, l, 0, 0);
#else
    *(v8h*)dst = *(const v8h*)src;
#endif
}
__device__ inline void lds_copy_wait() {
#if HAVE_ASYNC
    __builtin_amdgcn_s_wait_asynccnt(0);   // drain ASYNCcnt before barrier
#endif
}

// ---------------------------------------------------------------------------
// WMMA helpers (layouts per CDNA5 ISA 7.12.2)
// ---------------------------------------------------------------------------
__device__ inline v8f wmma16(v16h a, v16h b, v8f c) {
    // D = A(16x32 f16) * B(32x16 f16) + C(16x16 f32)
    return __builtin_amdgcn_wmma_f32_16x16x32_f16(
        false, a, false, b, (short)0, c, false, false);
}

// A fragment from a row-major [16 x K] tile (ld = row stride in halfs).
// Lane l: m = l&15, g = l>>4. Halves 0..15 = row[g*16 .. +15]
// (one contiguous 32B run -> two b128 loads).
__device__ inline v16h frag_a_rm(const half_t* __restrict__ base, int ld) {
    const int lane = threadIdx.x & 31;
    const int m = lane & 15, g = lane >> 4;
    const half_t* row = base + (size_t)m * ld + g * 16;
    v8h lo = *(const v8h*)row;
    v8h hi = *(const v8h*)(row + 8);
    v16h a;
#pragma unroll
    for (int i = 0; i < 8; ++i) { a[i] = lo[i]; a[i + 8] = hi[i]; }
    return a;
}

// ---------------------------------------------------------------------------
// Weight prep
// ---------------------------------------------------------------------------
__global__ __launch_bounds__(256) void convw_prep_kernel(
    const float* __restrict__ src, half_t* __restrict__ dst, int Cout, int Cin) {
    int i = blockIdx.x * 256 + threadIdx.x;
    int total = Cout * Cin * 9;
    if (i >= total) return;
    int k  = i % 9;
    int ci = (i / 9) % Cin;
    int co = i / (9 * Cin);
    dst[((size_t)k * Cout + co) * Cin + ci] = (half_t)src[i];   // [tap][Cout][Cin]
}

// dense f32 [K][N] -> f16 transposed [N][K] (B fragments become contiguous)
__global__ __launch_bounds__(256) void transpose_f16_kernel(
    const float* __restrict__ src, half_t* __restrict__ dst, int K, int N) {
    int i = blockIdx.x * 256 + threadIdx.x;
    if (i >= K * N) return;
    int k = i / N, n = i % N;
    dst[(size_t)n * K + k] = (half_t)src[i];
}

__global__ __launch_bounds__(256) void zero_f32_kernel(float* __restrict__ p, int count) {
    int i = blockIdx.x * 256 + threadIdx.x;
    if (i < count) p[i] = 0.0f;
}

__global__ __launch_bounds__(256) void fill_u32_kernel(unsigned* __restrict__ p,
                                                       unsigned v, int count) {
    int i = blockIdx.x * 256 + threadIdx.x;
    if (i < count) p[i] = v;
}

// ---------------------------------------------------------------------------
// Conv layer 0: Cin=3, 13x13, VALU f32 (K=27 too small for WMMA), fused
// BN+ReLU, 2x2 maxpool -> NHWC f16 [node][6][6][32].
// ---------------------------------------------------------------------------
__global__ __launch_bounds__(256) void conv0_kernel(
    const float* __restrict__ x, const float* __restrict__ w,
    const float* __restrict__ cb, const float* __restrict__ bg,
    const float* __restrict__ bb, half_t* __restrict__ out) {
    __shared__ float xs[3 * 169];
    __shared__ float wsm[32 * 27];
    __shared__ float act[169 * 32];
    const int tid = threadIdx.x;
    const size_t node = blockIdx.x;
    for (int i = tid; i < 3 * 169; i += 256) xs[i] = x[node * 507 + i];
    for (int i = tid; i < 32 * 27; i += 256) wsm[i] = w[i];
    __syncthreads();
    const float rs = rsqrtf(1.0f + 1e-5f);
    for (int idx = tid; idx < 32 * 169; idx += 256) {
        int c = idx / 169, p = idx % 169, py = p / 13, px = p % 13;
        float s = 0.0f;
#pragma unroll
        for (int ci = 0; ci < 3; ++ci)
#pragma unroll
            for (int ky = 0; ky < 3; ++ky) {
                int iy = py + ky - 1;
                if (iy < 0 || iy > 12) continue;
#pragma unroll
                for (int kx = 0; kx < 3; ++kx) {
                    int ix = px + kx - 1;
                    if (ix < 0 || ix > 12) continue;
                    s += xs[ci * 169 + iy * 13 + ix] * wsm[c * 27 + ci * 9 + ky * 3 + kx];
                }
            }
        float sc = bg[c] * rs;
        float y = fmaxf(s * sc + (cb[c] * sc + bb[c]), 0.0f);
        act[p * 32 + c] = y;
    }
    __syncthreads();
    for (int idx = tid; idx < 6 * 6 * 32; idx += 256) {
        int c = idx % 32, r = idx / 32, xo = r % 6, yo = r / 6;
        int p00 = (2 * yo) * 13 + 2 * xo;
        float mx = fmaxf(fmaxf(act[p00 * 32 + c], act[(p00 + 1) * 32 + c]),
                         fmaxf(act[(p00 + 13) * 32 + c], act[(p00 + 14) * 32 + c]));
        out[(node * 36 + r) * 32 + c] = (half_t)mx;
    }
}

// ---------------------------------------------------------------------------
// Conv layers 1..4 via WMMA: 9 shifted GEMMs (one per 3x3 tap).
// A = W[tap][Cout][Cin] (f16, global/L2), B = zero-padded input tile in LDS
// (NHWC f16, async-staged), C = [Cout x 144 positions] f32.
// Fused BN+ReLU (+2x2 maxpool).
// ---------------------------------------------------------------------------
template <int CIN, int COUT, int S, bool POOL>
__global__ __launch_bounds__(256) void conv_wmma_kernel(
    const half_t* __restrict__ actIn,    // [node][S][S][CIN]
    const half_t* __restrict__ w16,      // [9][COUT][CIN]
    const float* __restrict__ cbias, const float* __restrict__ bng,
    const float* __restrict__ bnb,
    half_t* __restrict__ actOut) {       // POOL ? [node][S/2][S/2][COUT] : [node][S][S][COUT]
    constexpr int NB   = (S == 6) ? 4 : 16;   // nodes per block -> N = 144
    constexpr int NPOS = NB * S * S;          // 144
    constexpr int NT   = NPOS / 16;           // 9 N-tiles
    constexpr int MT   = COUT / 16;
    constexpr int TOT  = MT * NT;
    constexpr int TPW  = (TOT + 7) / 8;       // tiles per wave (8 waves)
    constexpr int PS   = S + 2;               // padded spatial
    constexpr int XE   = NB * PS * PS * CIN;
    constexpr int YE   = POOL ? NPOS * COUT : 0;
    constexpr int SMH  = (XE > YE) ? XE : YE;
    __shared__ __align__(32) half_t smem[SMH];   // xpad, later reused for pool staging
    half_t* xpad = smem;
    half_t* yls  = smem;
    const int tid = threadIdx.x, wave = tid >> 5, lane = tid & 31;
    const int nodeBase = blockIdx.x * NB;

    // --- stage zero-padded input tile ---
    // 1) zero halo cells only (vector stores over channels)
    for (int idx = tid; idx < NB * PS * PS * (CIN / 8); idx += 256) {
        int cv = idx % (CIN / 8); int r = idx / (CIN / 8);
        int px = r % PS; r /= PS;
        int py = r % PS; int nb = r / PS;
        if (px == 0 || px == PS - 1 || py == 0 || py == PS - 1) {
            v8h z;
#pragma unroll
            for (int i = 0; i < 8; ++i) z[i] = (half_t)0.0f;
            *(v8h*)(xpad + ((size_t)(nb * PS + py) * PS + px) * CIN + cv * 8) = z;
        }
    }
    // 2) async-copy interior rows (each row = S*CIN contiguous halfs, both sides)
    constexpr int ROWH = S * CIN;            // 192 halfs for all layers
    constexpr int CHUNKS = ROWH / 8;         // 16B chunks per row
    for (int c = tid; c < NB * S * CHUNKS; c += 256) {
        int o = (c % CHUNKS) * 8; int r = c / CHUNKS;
        int y = r % S; int nb = r / S;
        const half_t* src = actIn + ((size_t)(nodeBase + nb) * S + y) * ROWH + o;
        half_t* dst = xpad + (((size_t)(nb * PS) + y + 1) * PS + 1) * CIN + o;
        lds_copy16(dst, src);
    }
    lds_copy_wait();
    __syncthreads();

    v8f acc[TPW];
    for (int s = 0; s < TPW; ++s)
        for (int v = 0; v < 8; ++v) acc[s][v] = 0.0f;

    for (int off = 0; off < 9; ++off) {
        const int dy = off / 3, dx = off % 3;
        const half_t* wp = w16 + (size_t)off * COUT * CIN;
#pragma unroll
        for (int kk = 0; kk < CIN / 32; ++kk) {
#pragma unroll
            for (int s = 0; s < TPW; ++s) {
                int t = s * 8 + wave;
                if (t >= TOT) continue;
                int mt = t / NT, ntl = t % NT;
                // A fragment: weight rows cout = mt*16..+15, K-slice kk*32..+31
                v16h af = frag_a_rm(wp + (size_t)mt * 16 * CIN + kk * 32, CIN);
                // B fragment: 16 contiguous halfs (32B vector LDS load) per lane
                int nloc = ntl * 16 + (lane & 15);
                int pos = nloc % (S * S), nb = nloc / (S * S);
                int py = pos / S + dy, px = pos % S + dx;
                const v16h* bp = (const v16h*)(xpad + (((nb * PS + py) * PS) + px) * CIN
                                               + kk * 32 + (lane >> 4) * 16);
                v16h bf = *bp;
                acc[s] = wmma16(af, bf, acc[s]);
            }
        }
    }

    const int g = lane >> 4;
    const float rs = rsqrtf(1.0f + 1e-5f);
    if (!POOL) {
#pragma unroll
        for (int s = 0; s < TPW; ++s) {
            int t = s * 8 + wave;
            if (t >= TOT) continue;
            int mt = t / NT, ntl = t % NT;
            int nloc = ntl * 16 + (lane & 15);
            int pos = nloc % (S * S), nb = nloc / (S * S);
            size_t node = nodeBase + nb;
#pragma unroll
            for (int v = 0; v < 8; ++v) {
                int co = mt * 16 + v + 8 * g;
                float sc = bng[co] * rs;
                float y = fmaxf(acc[s][v] * sc + (cbias[co] * sc + bnb[co]), 0.0f);
                actOut[((size_t)node * S * S + pos) * COUT + co] = (half_t)y;
            }
        }
    } else {
        __syncthreads();   // all xpad reads done; reuse smem for pooling
#pragma unroll
        for (int s = 0; s < TPW; ++s) {
            int t = s * 8 + wave;
            if (t >= TOT) continue;
            int mt = t / NT, ntl = t % NT;
            int nloc = ntl * 16 + (lane & 15);
#pragma unroll
            for (int v = 0; v < 8; ++v) {
                int co = mt * 16 + v + 8 * g;
                float sc = bng[co] * rs;
                float y = fmaxf(acc[s][v] * sc + (cbias[co] * sc + bnb[co]), 0.0f);
                yls[(size_t)nloc * COUT + co] = (half_t)y;
            }
        }
        __syncthreads();
        constexpr int SO = S / 2;
        for (int idx = tid; idx < NB * SO * SO * COUT; idx += 256) {
            int co = idx % COUT; int r = idx / COUT;
            int xo = r % SO; r /= SO;
            int yo = r % SO; int nb = r / SO;
            int p00 = (nb * S + 2 * yo) * S + 2 * xo;
            float mx = fmaxf(
                fmaxf((float)yls[(size_t)p00 * COUT + co],
                      (float)yls[(size_t)(p00 + 1) * COUT + co]),
                fmaxf((float)yls[(size_t)(p00 + S) * COUT + co],
                      (float)yls[(size_t)(p00 + S + 1) * COUT + co]));
            size_t node = nodeBase + nb;
            actOut[((size_t)node * SO * SO + (size_t)yo * SO + xo) * COUT + co] = (half_t)mx;
        }
    }
}

// ---------------------------------------------------------------------------
// Tiled WMMA GEMM: C[M,N] = A[M,K] x Bt^T, A row-major f16 (async-staged in
// LDS, reused by 4 N-tile waves), Bt = weights pre-transposed [N][K] so each
// lane's B fragment is one contiguous 32B global load (L2-resident weights).
// ---------------------------------------------------------------------------
template <int OUT16, int BIAS, int RELU>
__global__ __launch_bounds__(256) void gemm_wmma_kernel(
    const half_t* __restrict__ A, const half_t* __restrict__ Bt,
    const float* __restrict__ bias, void* __restrict__ Cout,
    int M, int Nn, int K) {
    __shared__ __align__(16) half_t As[64 * 40];
    const int tid = threadIdx.x;
    const int wave = tid >> 5, lane = tid & 31;
    const int m0 = blockIdx.x * 64, n0 = blockIdx.y * 64;
    const int nt = wave & 3;
    const int mt0 = (wave >> 2) * 2, mt1 = mt0 + 1;
    v8f acc0, acc1;
    for (int v = 0; v < 8; ++v) { acc0[v] = 0.0f; acc1[v] = 0.0f; }

    const int arow = tid >> 2, acol = (tid & 3) * 8;   // 64 rows x 32 cols, 16B chunks
    int gm = m0 + arow;
    if (gm >= M) gm = M - 1;   // clamp: garbage rows land in never-stored C rows

    const int n = lane & 15, g = lane >> 4;
    const half_t* brow = Bt + (size_t)(n0 + nt * 16 + n) * K + g * 16;

    for (int k0 = 0; k0 < K; k0 += 32) {
        lds_copy16(As + arow * 40 + acol, A + (size_t)gm * K + k0 + acol);
        if (k0 + 32 < K)   // pull next K tile toward L2/WGP$ (global_prefetch_b8)
            __builtin_prefetch(A + (size_t)gm * K + k0 + 32 + acol, 0, 0);
        lds_copy_wait();
        __syncthreads();
        v16h bf = *(const v16h*)(brow + k0);
        v16h a0 = frag_a_rm(As + mt0 * 16 * 40, 40);
        v16h a1 = frag_a_rm(As + mt1 * 16 * 40, 40);
        acc0 = wmma16(a0, bf, acc0);
        acc1 = wmma16(a1, bf, acc1);
        __syncthreads();
    }

    const int gn = n0 + nt * 16 + n;
    const float bv = BIAS ? bias[gn] : 0.0f;
#pragma unroll
    for (int v = 0; v < 8; ++v) {
        int gm0 = m0 + mt0 * 16 + v + 8 * g;
        int gm1 = m0 + mt1 * 16 + v + 8 * g;
        float y0 = acc0[v] + bv, y1 = acc1[v] + bv;
        if (RELU) { y0 = fmaxf(y0, 0.0f); y1 = fmaxf(y1, 0.0f); }
        if (OUT16) {
            half_t* C = (half_t*)Cout;
            if (gm0 < M) C[(size_t)gm0 * Nn + gn] = (half_t)y0;
            if (gm1 < M) C[(size_t)gm1 * Nn + gn] = (half_t)y1;
        } else {
            float* C = (float*)Cout;
            if (gm0 < M) C[(size_t)gm0 * Nn + gn] = y0;
            if (gm1 < M) C[(size_t)gm1 * Nn + gn] = y1;
        }
    }
}

// ---------------------------------------------------------------------------
// GAT edge machinery
// ---------------------------------------------------------------------------
__device__ inline unsigned f2ord(float f) {
    unsigned u = __float_as_uint(f);
    return (u & 0x80000000u) ? ~u : (u | 0x80000000u);
}
__device__ inline float ord2f(unsigned u) {
    u = (u & 0x80000000u) ? (u & 0x7fffffffu) : ~u;
    return __uint_as_float(u);
}

__global__ __launch_bounds__(256) void alpha_dot_kernel(
    const float* __restrict__ h, const float* __restrict__ av_src,
    const float* __restrict__ av_dst, float* __restrict__ asrc,
    float* __restrict__ adst) {
    int i = blockIdx.x * 256 + threadIdx.x;
    if (i >= NNODES * 4) return;
    int nidx = i >> 2, hd = i & 3;
    const float* hp = h + (size_t)nidx * 512 + hd * 128;
    const float* sv = av_src + hd * 128;
    const float* dv = av_dst + hd * 128;
    float s1 = 0.0f, s2 = 0.0f;
    for (int c = 0; c < 128; ++c) { float x = hp[c]; s1 += x * sv[c]; s2 += x * dv[c]; }
    asrc[i] = s1;
    adst[i] = s2;
}

__device__ inline void edge_sd(const int* __restrict__ ei, int e, int& s, int& d) {
    if (e < NEDGES) { s = ei[e]; d = ei[NEDGES + e]; }
    else            { s = d = e - NEDGES; }   // self loops
}

__global__ __launch_bounds__(256) void edge_logits_max_kernel(
    const int* __restrict__ ei, const float* __restrict__ asrc,
    const float* __restrict__ adst, float* __restrict__ ealpha,
    unsigned* __restrict__ segmax) {
    int e = blockIdx.x * 256 + threadIdx.x;
    if (e >= EAUG) return;
    int s, d; edge_sd(ei, e, s, d);
#pragma unroll
    for (int hd = 0; hd < 4; ++hd) {
        float l = asrc[s * 4 + hd] + adst[d * 4 + hd];
        l = (l > 0.0f) ? l : 0.2f * l;          // leaky_relu(0.2)
        ealpha[(size_t)e * 4 + hd] = l;
        atomicMax(&segmax[d * 4 + hd], f2ord(l));
    }
}

__global__ __launch_bounds__(256) void edge_exp_sum_kernel(
    const int* __restrict__ ei, float* __restrict__ ealpha,
    const unsigned* __restrict__ segmax, float* __restrict__ segsum) {
    int e = blockIdx.x * 256 + threadIdx.x;
    if (e >= EAUG) return;
    int s, d; edge_sd(ei, e, s, d);
    (void)s;
#pragma unroll
    for (int hd = 0; hd < 4; ++hd) {
        float l = ealpha[(size_t)e * 4 + hd];
        float ex = __expf(l - ord2f(segmax[d * 4 + hd]));
        ealpha[(size_t)e * 4 + hd] = ex;
        atomicAdd(&segsum[d * 4 + hd], ex);
    }
}

// one wave per edge: scatter 512 channels of h[src]*alpha into agg[dst]
__global__ __launch_bounds__(256) void edge_scatter_kernel(
    const int* __restrict__ ei, const float* __restrict__ ealpha,
    const float* __restrict__ segsum, const float* __restrict__ h,
    float* __restrict__ agg) {
    int e = blockIdx.x * 8 + (threadIdx.x >> 5);
    if (e >= EAUG) return;
    const int lane = threadIdx.x & 31;
    int s, d; edge_sd(ei, e, s, d);
    float wt[4];
#pragma unroll
    for (int hd = 0; hd < 4; ++hd)
        wt[hd] = ealpha[(size_t)e * 4 + hd] / (segsum[d * 4 + hd] + 1e-16f);
    const float* hs = h + (size_t)s * 512;
    float* ag = agg + (size_t)d * 512;
    for (int j = lane; j < 512; j += 32)
        atomicAdd(&ag[j], hs[j] * wt[j >> 7]);
}

__global__ __launch_bounds__(256) void bias_relu_f16_kernel(
    const float* __restrict__ agg, const float* __restrict__ bias,
    half_t* __restrict__ out, int count) {
    int i = blockIdx.x * 256 + threadIdx.x;
    if (i >= count) return;
    float v = agg[i] + bias[i & 511];
    out[i] = (half_t)fmaxf(v, 0.0f);
}

// final 128 -> 5 head (tiny; VALU)
__global__ __launch_bounds__(256) void fc1_kernel(
    const half_t* __restrict__ fc0, const float* __restrict__ w1,
    const float* __restrict__ b1, float* __restrict__ out) {
    __shared__ float wsh[128 * 5];
    __shared__ float bsh[5];
    const int tid = threadIdx.x;
    for (int i = tid; i < 128 * 5; i += 256) wsh[i] = w1[i];
    if (tid < 5) bsh[tid] = b1[tid];
    __syncthreads();
    int nidx = blockIdx.x * 256 + tid;
    if (nidx >= NNODES) return;
    const half_t* f = fc0 + (size_t)nidx * 128;
    float acc[5];
#pragma unroll
    for (int k = 0; k < 5; ++k) acc[k] = bsh[k];
    for (int c = 0; c < 128; ++c) {
        float v = (float)f[c];
#pragma unroll
        for (int k = 0; k < 5; ++k) acc[k] += v * wsh[c * 5 + k];
    }
#pragma unroll
    for (int k = 0; k < 5; ++k) out[(size_t)nidx * 5 + k] = acc[k];
}

// ---------------------------------------------------------------------------
// Host-side orchestration
// ---------------------------------------------------------------------------
static inline dim3 g1(long long n, int b) { return dim3((unsigned)((n + b - 1) / b)); }

extern "C" void kernel_launch(void* const* d_in, const int* in_sizes, int n_in,
                              void* d_out, int out_size, void* d_ws, size_t ws_size,
                              hipStream_t stream) {
    (void)in_sizes; (void)n_in; (void)out_size; (void)ws_size;
    const float* x  = (const float*)d_in[0];
    const int*   ei = (const int*)d_in[1];
    const float* conv_w[5] = {(const float*)d_in[2],  (const float*)d_in[6],
                              (const float*)d_in[10], (const float*)d_in[14],
                              (const float*)d_in[18]};
    const float* conv_b[5] = {(const float*)d_in[3],  (const float*)d_in[7],
                              (const float*)d_in[11], (const float*)d_in[15],
                              (const float*)d_in[19]};
    const float* bn_g[5]   = {(const float*)d_in[4],  (const float*)d_in[8],
                              (const float*)d_in[12], (const float*)d_in[16],
                              (const float*)d_in[20]};
    const float* bn_b[5]   = {(const float*)d_in[5],  (const float*)d_in[9],
                              (const float*)d_in[13], (const float*)d_in[17],
                              (const float*)d_in[21]};
    const float* mlp_w = (const float*)d_in[22];
    const float* mlp_b = (const float*)d_in[23];
    const float* gat_w[2]    = {(const float*)d_in[24], (const float*)d_in[28]};
    const float* gat_asrc[2] = {(const float*)d_in[25], (const float*)d_in[29]};
    const float* gat_adst[2] = {(const float*)d_in[26], (const float*)d_in[30]};
    const float* gat_bias[2] = {(const float*)d_in[27], (const float*)d_in[31]};
    const float* fc_w0 = (const float*)d_in[32];
    const float* fc_b0 = (const float*)d_in[33];
    const float* fc_w1 = (const float*)d_in[34];
    const float* fc_b1 = (const float*)d_in[35];

    char* ws = (char*)d_ws;

    // f16 weight pool (offsets in halfs); dense weights stored transposed [N][K]
    const size_t HW_C1  = 0;                       // [9][32][32]
    const size_t HW_C2  = HW_C1 + 9 * 32 * 32;     // [9][64][32]
    const size_t HW_C3  = HW_C2 + 9 * 64 * 32;     // [9][64][64]
    const size_t HW_C4  = HW_C3 + 9 * 64 * 64;     // [9][128][64]
    const size_t HW_MLP = HW_C4 + 9 * 128 * 64;    // [128][128]^T
    const size_t HW_G0  = HW_MLP + 128 * 128;      // [512][128]  (gat_w0^T)
    const size_t HW_G1  = HW_G0 + 128 * 512;       // [512][512]^T
    const size_t HW_F0  = HW_G1 + 512 * 512;       // [128][512]  (fc_w0^T)
    const size_t HW_END = HW_F0 + 512 * 128;       // 547840 halfs

    // byte layout (256-aligned regions, lifetime-based overlays)
    const size_t B_ACT  = (size_t)NNODES * 36 * 32 * 2;   // 46.08 MB (conv ping/pong)
    const size_t B_H32  = (size_t)NNODES * 512 * 4;       // 40.96 MB
    const size_t B_HIN  = (size_t)NNODES * 512 * 2;       // 20.48 MB
    const size_t oW16   = 0;
    const size_t oR0    = (HW_END * 2 + 255) & ~(size_t)255;
    const size_t R0_SZ  = 2 * B_H32 + B_HIN;              // 102.4 MB >= 2*B_ACT
    const size_t oFeat4 = oR0 + R0_SZ;
    const size_t oMlp16 = oFeat4 + (size_t)NNODES * 128 * 2;
    const size_t oFc016 = oMlp16 + (size_t)NNODES * 128 * 2;
    const size_t oAsrc  = oFc016 + (size_t)NNODES * 128 * 2;
    const size_t oAdst  = oAsrc + (size_t)NNODES * 4 * 4;
    const size_t oSegMx = oAdst + (size_t)NNODES * 4 * 4;
    const size_t oSegSm = oSegMx + (size_t)NNODES * 4 * 4;
    const size_t oEalph = oSegSm + (size_t)NNODES * 4 * 4;

    half_t*  w16    = (half_t*)(ws + oW16);
    half_t*  actA   = (half_t*)(ws + oR0);
    half_t*  actB   = (half_t*)(ws + oR0 + B_ACT);
    float*   h32    = (float*)(ws + oR0);                 // overlays actA/actB (convs done)
    float*   agg32  = (float*)(ws + oR0 + B_H32);
    half_t*  hin16  = (half_t*)(ws + oR0 + 2 * B_H32);
    half_t*  feat4  = (half_t*)(ws + oFeat4);
    half_t*  mlp16  = (half_t*)(ws + oMlp16);
    half_t*  fc016  = (half_t*)(ws + oFc016);
    float*   asrc   = (float*)(ws + oAsrc);
    float*   adst   = (float*)(ws + oAdst);
    unsigned* segmx = (unsigned*)(ws + oSegMx);
    float*   segsm  = (float*)(ws + oSegSm);
    float*   ealpha = (float*)(ws + oEalph);

    // ---- weight prep ----
    convw_prep_kernel<<<g1(32 * 32 * 9, 256), 256, 0, stream>>>(conv_w[1], w16 + HW_C1, 32, 32);
    convw_prep_kernel<<<g1(64 * 32 * 9, 256), 256, 0, stream>>>(conv_w[2], w16 + HW_C2, 64, 32);
    convw_prep_kernel<<<g1(64 * 64 * 9, 256), 256, 0, stream>>>(conv_w[3], w16 + HW_C3, 64, 64);
    convw_prep_kernel<<<g1(128 * 64 * 9, 256), 256, 0, stream>>>(conv_w[4], w16 + HW_C4, 128, 64);
    transpose_f16_kernel<<<g1(128 * 128, 256), 256, 0, stream>>>(mlp_w,    w16 + HW_MLP, 128, 128);
    transpose_f16_kernel<<<g1(128 * 512, 256), 256, 0, stream>>>(gat_w[0], w16 + HW_G0,  128, 512);
    transpose_f16_kernel<<<g1(512 * 512, 256), 256, 0, stream>>>(gat_w[1], w16 + HW_G1,  512, 512);
    transpose_f16_kernel<<<g1(512 * 128, 256), 256, 0, stream>>>(fc_w0,    w16 + HW_F0,  512, 128);

    // ---- CNN ----
    conv0_kernel<<<NNODES, 256, 0, stream>>>(x, conv_w[0], conv_b[0], bn_g[0], bn_b[0], actA);
    conv_wmma_kernel<32, 32, 6, false><<<NNODES / 4, 256, 0, stream>>>(
        actA, w16 + HW_C1, conv_b[1], bn_g[1], bn_b[1], actB);
    conv_wmma_kernel<32, 64, 6, true><<<NNODES / 4, 256, 0, stream>>>(
        actB, w16 + HW_C2, conv_b[2], bn_g[2], bn_b[2], actA);
    conv_wmma_kernel<64, 64, 3, false><<<NNODES / 16, 256, 0, stream>>>(
        actA, w16 + HW_C3, conv_b[3], bn_g[3], bn_b[3], actB);
    conv_wmma_kernel<64, 128, 3, true><<<NNODES / 16, 256, 0, stream>>>(
        actB, w16 + HW_C4, conv_b[4], bn_g[4], bn_b[4], feat4);

    // ---- compress MLP: [N,128] x [128,128] + bias, relu -> f16 ----
    gemm_wmma_kernel<1, 1, 1><<<dim3(313, 2), 256, 0, stream>>>(
        feat4, w16 + HW_MLP, mlp_b, mlp16, NNODES, 128, 128);

    // ---- two GAT layers ----
    for (int l = 0; l < 2; ++l) {
        const half_t* Ain = (l == 0) ? mlp16 : hin16;
        const int K = (l == 0) ? 128 : 512;
        gemm_wmma_kernel<0, 0, 0><<<dim3(313, 8), 256, 0, stream>>>(
            Ain, w16 + ((l == 0) ? HW_G0 : HW_G1), nullptr, h32, NNODES, 512, K);
        alpha_dot_kernel<<<g1((long long)NNODES * 4, 256), 256, 0, stream>>>(
            h32, gat_asrc[l], gat_adst[l], asrc, adst);
        fill_u32_kernel<<<g1((long long)NNODES * 4, 256), 256, 0, stream>>>(
            segmx, 0x007FFFFFu /* f2ord(-inf) */, NNODES * 4);
        zero_f32_kernel<<<g1((long long)NNODES * 4, 256), 256, 0, stream>>>(segsm, NNODES * 4);
        zero_f32_kernel<<<g1((long long)NNODES * 512, 256), 256, 0, stream>>>(
            agg32, NNODES * 512);
        edge_logits_max_kernel<<<g1(EAUG, 256), 256, 0, stream>>>(ei, asrc, adst, ealpha, segmx);
        edge_exp_sum_kernel<<<g1(EAUG, 256), 256, 0, stream>>>(ei, ealpha, segmx, segsm);
        edge_scatter_kernel<<<g1(EAUG, 8), 256, 0, stream>>>(ei, ealpha, segsm, h32, agg32);
        bias_relu_f16_kernel<<<g1((long long)NNODES * 512, 256), 256, 0, stream>>>(
            agg32, gat_bias[l], hin16, NNODES * 512);
    }

    // ---- actions MLP: 512 -> 128 (relu) -> 5 ----
    gemm_wmma_kernel<1, 1, 1><<<dim3(313, 2), 256, 0, stream>>>(
        hin16, w16 + HW_F0, fc_b0, fc016, NNODES, 128, 512);
    fc1_kernel<<<g1(NNODES, 256), 256, 0, stream>>>(fc016, fc_w1, fc_b1, (float*)d_out);
}
